// Attention_10565619548720
// MI455X (gfx1250) — matrix operand order repaired and checked by reading, not compiled, not run on hardware
//
#include <hip/hip_runtime.h>
#include <hip/hip_bf16.h>

// ---------------- CDNA5 WMMA types & fragment helpers ----------------
typedef __attribute__((ext_vector_type(16))) _Float16 v16h;
typedef __attribute__((ext_vector_type(8)))  _Float16 v8h;
typedef __attribute__((ext_vector_type(8)))  float    v8f;

// A-matrix 16x32 f16 fragment from row-major tile (ld elements per row).
__device__ __forceinline__ v16h frag_a_load(const _Float16* __restrict__ tile, int ld) {
  const int lane = threadIdx.x & 31;
  const int m = lane & 15, half = lane >> 4;
  const _Float16* row = tile + m * ld + half * 16;
  v16h f;
#pragma unroll
  for (int p = 0; p < 8; ++p) {
    const int k = 2 * p; // 0,2,4,...,14
    f[2 * p]     = row[k];
    f[2 * p + 1] = row[k + 1];
  }
  return f;
}

// B-matrix 32x16 f16 fragment from a TRANSPOSED tile laid out [n][k] (ld = k-stride).
// Lane n holds K = 16*half + 0..15 contiguous -> 2 x b128 per lane.
__device__ __forceinline__ v16h frag_b_load(const _Float16* __restrict__ tileT, int ld) {
  const int lane = threadIdx.x & 31;
  const int n = lane & 15, half = lane >> 4;
  const _Float16* src = tileT + n * ld + half * 16;
  v16h f;
#pragma unroll
  for (int e = 0; e < 16; ++e) f[e] = src[e];
  return f;
}

__device__ __forceinline__ v8f wmma_f16(v16h a, v16h b, v8f c) {
  return __builtin_amdgcn_wmma_f32_16x16x32_f16(false, a, false, b, (short)0, c, false, false);
}

// Async global->LDS copy, 16B per lane, tracked with ASYNCcnt (CDNA5 TDM-class path).
__device__ __forceinline__ void async_load_b128(unsigned lds_off, unsigned long long gaddr) {
  asm volatile("global_load_async_to_lds_b128 %0, %1, off"
               :: "v"(lds_off), "v"(gaddr) : "memory");
}
__device__ __forceinline__ void wait_async4() {
  asm volatile("s_wait_asynccnt 0x4" ::: "memory");
}
__device__ __forceinline__ void wait_async0() {
  asm volatile("s_wait_asynccnt 0x0" ::: "memory");
}

// ---------------- fp32 -> f16 conversion kernels ----------------
__global__ __launch_bounds__(256) void cvt_mat_kernel(const float* __restrict__ src,
    _Float16* __restrict__ dst, int rows, int cols, int dstLd, int colOff) {
  size_t idx = (size_t)blockIdx.x * blockDim.x + threadIdx.x;
  if (idx >= (size_t)rows * cols) return;
  int r = (int)(idx / cols), c = (int)(idx % cols);
  dst[(size_t)r * dstLd + colOff + c] = (_Float16)src[idx];
}

// Tiled f32->f16 transpose: src K x N (row-major) -> dst N x K (row-major).
// Both global sides coalesced; LDS tile 64x64 padded.
__global__ __launch_bounds__(256) void cvt_transpose_kernel(const float* __restrict__ src,
    _Float16* __restrict__ dst, int K, int N) {
  __shared__ _Float16 sT[64][72];
  const int k0 = blockIdx.y * 64, n0 = blockIdx.x * 64;
  const int t = threadIdx.x;
  const int r = t >> 2, c4 = (t & 3) * 16;
  const float* sp = src + (size_t)(k0 + r) * N + n0 + c4;
#pragma unroll
  for (int j = 0; j < 16; ++j) sT[c4 + j][r] = (_Float16)sp[j];
  __syncthreads();
  _Float16* dp = dst + (size_t)(n0 + r) * K + k0 + c4;
#pragma unroll
  for (int j = 0; j < 16; ++j) dp[j] = sT[r][c4 + j];
}

// ---------------- WMMA GEMM: C(f32, MxN) = A(f16, MxK) * Bt(f16, NxK)^T ----------------
// Block tile 128x128, 8 waves (2x4), K-step 32, double-buffered LDS fed by async loads.
__global__ __launch_bounds__(256) void gemm_f16_kernel(const _Float16* __restrict__ A,
    const _Float16* __restrict__ Bt, float* __restrict__ C, int M, int N, int K) {
  __shared__ _Float16 sA[2][128][40];   // 80B rows: 16B-aligned, conflict-free b32 frags
  __shared__ _Float16 sBt[2][128][56];  // 112B rows: 16B-aligned, conflict-free b128 frags
  const int tid = threadIdx.x;
  const int wid = tid >> 5, lane = tid & 31;
  const int wm = (wid >> 2) * 64, wn = (wid & 3) * 32;
  const int bm = blockIdx.y * 128, bn = blockIdx.x * 128;

  // Staging geometry: 16B chunks; chunk c -> row = c>>2, off = (c&3)*8 elements.
  // Per wave: 2 A-chunks + 2 B-chunks per tile = 4 async instructions.
  unsigned long long ga[2], gb[2];
  unsigned la[2][2], lb[2][2];
#pragma unroll
  for (int j = 0; j < 2; ++j) {
    const int c = wid * 64 + j * 32 + lane;
    const int r = c >> 2, o = (c & 3) * 8;
    ga[j] = (unsigned long long)(A  + (size_t)(bm + r) * K + o);
    gb[j] = (unsigned long long)(Bt + (size_t)(bn + r) * K + o);
    la[0][j] = (unsigned)(size_t)&sA[0][r][o];
    la[1][j] = (unsigned)(size_t)&sA[1][r][o];
    lb[0][j] = (unsigned)(size_t)&sBt[0][r][o];
    lb[1][j] = (unsigned)(size_t)&sBt[1][r][o];
  }

  v8f acc[4][2];
#pragma unroll
  for (int i = 0; i < 4; ++i)
#pragma unroll
    for (int j = 0; j < 2; ++j)
#pragma unroll
      for (int e = 0; e < 8; ++e) acc[i][j][e] = 0.0f;

  const int nt = K >> 5;
  // prologue: stage tile 0 into buffer 0
#pragma unroll
  for (int j = 0; j < 2; ++j) {
    async_load_b128(la[0][j], ga[j]);
    async_load_b128(lb[0][j], gb[j]);
  }

  for (int it = 0; it < nt; ++it) {
    const int buf = it & 1;
    if (it) __syncthreads();                 // WAR: everyone done reading buf^1
    if (it + 1 < nt) {                       // stage next tile into buf^1
      const unsigned long long koff = (unsigned long long)(it + 1) * 64; // 32 f16
#pragma unroll
      for (int j = 0; j < 2; ++j) {
        async_load_b128(la[buf ^ 1][j], ga[j] + koff);
        async_load_b128(lb[buf ^ 1][j], gb[j] + koff);
      }
      wait_async4();                         // current tile's 4 asyncs complete
    } else {
      wait_async0();
    }
    __syncthreads();                         // all waves' staging visible

    v16h af[4];
#pragma unroll
    for (int mt = 0; mt < 4; ++mt) af[mt] = frag_a_load(&sA[buf][wm + mt * 16][0], 40);
#pragma unroll
    for (int nb = 0; nb < 2; ++nb) {
      v16h bf = frag_b_load(&sBt[buf][wn + nb * 16][0], 56);
#pragma unroll
      for (int mt = 0; mt < 4; ++mt) acc[mt][nb] = wmma_f16(af[mt], bf, acc[mt][nb]);
    }
  }

  // epilogue: uniform base + 32-bit divergent offsets (SADDR + voffset form)
  const int cn = lane & 15, half = lane >> 4;
#pragma unroll
  for (int mt = 0; mt < 4; ++mt)
#pragma unroll
    for (int nb = 0; nb < 2; ++nb) {
      float* dst = C + (size_t)(bm + wm + mt * 16) * N + (bn + wn + nb * 16);
      const int off = 8 * half * N + cn;
#pragma unroll
      for (int r = 0; r < 8; ++r) dst[off + r * N] = acc[mt][nb][r];
    }
}

// ---------------- RoPE + cache write + f16 staging ----------------
__global__ __launch_bounds__(256) void rope_q_kernel(const float* __restrict__ qkv,
    const float* __restrict__ fcos, const float* __restrict__ fsin, _Float16* __restrict__ qout) {
  int idx = blockIdx.x * blockDim.x + threadIdx.x;   // B*S*H*64
  int d2 = idx & 63; int t = idx >> 6;
  int h = t & 31; t >>= 5;
  int s = t & 2047; int b = t >> 11;
  const float* src = qkv + ((size_t)(b * 2048 + s)) * 6144 + h * 128 + 2 * d2;
  float c = fcos[s * 64 + d2], sn = fsin[s * 64 + d2];
  float v0 = src[0], v1 = src[1];
  _Float16* dst = qout + ((size_t)((b * 32 + h) * 2048 + s)) * 128 + 2 * d2; // (B,H,S,HD)
  dst[0] = (_Float16)(v0 * c - v1 * sn);
  dst[1] = (_Float16)(v0 * sn + v1 * c);
}

__global__ __launch_bounds__(256) void rope_kv_kernel(const float* __restrict__ qkv,
    const float* __restrict__ fcos, const float* __restrict__ fsin,
    float* __restrict__ cache_k, float* __restrict__ cache_v,
    _Float16* __restrict__ kout, _Float16* __restrict__ vout) {
  int idx = blockIdx.x * blockDim.x + threadIdx.x;   // B*S*KVH*64
  int d2 = idx & 63; int t = idx >> 6;
  int kv = t & 7; t >>= 3;
  int s = t & 2047; int b = t >> 11;
  const float* row = qkv + ((size_t)(b * 2048 + s)) * 6144;
  const float* ks = row + 4096 + kv * 128 + 2 * d2;
  const float* vs = row + 5120 + kv * 128 + 2 * d2;
  float c = fcos[s * 64 + d2], sn = fsin[s * 64 + d2];
  float k0 = ks[0], k1 = ks[1];
  float r0 = k0 * c - k1 * sn, r1 = k0 * sn + k1 * c;
  size_t cidx = ((size_t)(b * 2048 + s) * 8 + kv) * 128 + 2 * d2;     // (B,S,KVH,HD)
  cache_k[cidx] = r0; cache_k[cidx + 1] = r1;
  float v0 = vs[0], v1 = vs[1];
  cache_v[cidx] = v0; cache_v[cidx + 1] = v1;
  size_t fidx = ((size_t)((b * 8 + kv) * 2048 + s)) * 128 + 2 * d2;   // (B,KVH,S,HD)
  kout[fidx] = (_Float16)r0; kout[fidx + 1] = (_Float16)r1;
  vout[fidx] = (_Float16)v0; vout[fidx + 1] = (_Float16)v1;
}

// ---------------- Flash attention (causal, GQA) ----------------
// Grid: (S/64, H, B); 4 waves, each owns 16 query rows; 64-key tiles via WMMA.
__global__ __launch_bounds__(128) void attn_kernel(const _Float16* __restrict__ Q,
    const _Float16* __restrict__ Kc, const _Float16* __restrict__ Vc,
    _Float16* __restrict__ O) {
  __shared__ _Float16 sK[64][136];     // [key][d], conflict-free b128 frag reads
  __shared__ _Float16 sVt[128][72];    // [d][key]
  __shared__ _Float16 sP[4][16][72];   // per-wave P tile (A-fragment staging)
  const int qt = blockIdx.x, h = blockIdx.y, b = blockIdx.z;
  const int kvh = h >> 2;              // N_REP = 4
  const int tid = threadIdx.x, w = tid >> 5, lane = tid & 31;
  const int ncol = lane & 15, half = lane >> 4;
  const int qbase = qt * 64 + w * 16;

  const _Float16* qptr = Q + (((size_t)b * 32 + h) * 2048 + qbase) * 128;
  v16h qf[4];
#pragma unroll
  for (int dc = 0; dc < 4; ++dc) qf[dc] = frag_a_load(qptr + dc * 32, 128);

  v8f acc[8];
  float rmax[8], rsum[8];
#pragma unroll
  for (int dt = 0; dt < 8; ++dt)
#pragma unroll
    for (int e = 0; e < 8; ++e) acc[dt][e] = 0.0f;
#pragma unroll
  for (int r = 0; r < 8; ++r) { rmax[r] = -3.0e38f; rsum[r] = 0.0f; }

  const _Float16* kbase = Kc + (((size_t)b * 8 + kvh) * 2048) * 128;
  const _Float16* vbase = Vc + (((size_t)b * 8 + kvh) * 2048) * 128;
  const int ntiles = qt + 1;
  const float scale = 0.08838834764831845f; // 1/sqrt(128)

  for (int it = 0; it < ntiles; ++it) {
    const int kt0 = it * 64;
    __syncthreads();
    // stage K tile via async global->LDS (4 instructions per wave)
    {
      const _Float16* gk = kbase + (size_t)kt0 * 128;
#pragma unroll
      for (int j = 0; j < 4; ++j) {
        const int c = j * 128 + tid;            // 512 x 16B chunks
        const int row = c >> 3, o = (c & 7) * 8;
        async_load_b128((unsigned)(size_t)&sK[row][o],
                        (unsigned long long)(gk + (size_t)row * 128 + o));
      }
    }
    // stage V tile transposed, packing two keys per b32 LDS write
    for (int i = tid; i < 64 * 32; i += 128) {
      int d2 = i & 63, kp = i >> 6;
      const unsigned* r0 = (const unsigned*)(vbase + (size_t)(kt0 + 2 * kp) * 128);
      const unsigned* r1 = (const unsigned*)(vbase + (size_t)(kt0 + 2 * kp + 1) * 128);
      unsigned a = r0[d2], bb = r1[d2];
      *(unsigned*)&sVt[2 * d2][2 * kp]     = (a & 0xffffu) | (bb << 16);
      *(unsigned*)&sVt[2 * d2 + 1][2 * kp] = (a >> 16) | (bb & 0xffff0000u);
    }
    wait_async0();
    __syncthreads();

    // S = Q K^T : 16 WMMAs
    v8f sc[4];
#pragma unroll
    for (int nb = 0; nb < 4; ++nb) {
      v8f t;
#pragma unroll
      for (int e = 0; e < 8; ++e) t[e] = 0.0f;
#pragma unroll
      for (int dc = 0; dc < 4; ++dc)
        t = wmma_f16(qf[dc], frag_b_load(&sK[nb * 16][dc * 32], 136), t);
      sc[nb] = t;
    }

    const bool lastTile = (it == ntiles - 1);
    float fac[8];
#pragma unroll
    for (int r = 0; r < 8; ++r) {
      const int row = qbase + r + 8 * half;
      float m = -3.0e38f;
#pragma unroll
      for (int nb = 0; nb < 4; ++nb) {
        float v = sc[nb][r] * scale;
        if (lastTile && (kt0 + nb * 16 + ncol) > row) v += -1.0e9f;
        sc[nb][r] = v;
        m = fmaxf(m, v);
      }
#pragma unroll
      for (int xm = 1; xm < 16; xm <<= 1) m = fmaxf(m, __shfl_xor(m, xm, 32));
      float nm = fmaxf(rmax[r], m);
      fac[r] = __expf(rmax[r] - nm);
      rmax[r] = nm;
      float ps = 0.0f;
#pragma unroll
      for (int nb = 0; nb < 4; ++nb) {
        float p = __expf(sc[nb][r] - nm);
        ps += p;
        sP[w][r + 8 * half][nb * 16 + ncol] = (_Float16)p;
      }
#pragma unroll
      for (int xm = 1; xm < 16; xm <<= 1) ps += __shfl_xor(ps, xm, 32);
      rsum[r] = rsum[r] * fac[r] + ps;
    }
#pragma unroll
    for (int dt = 0; dt < 8; ++dt)
#pragma unroll
      for (int r = 0; r < 8; ++r) acc[dt][r] *= fac[r];

    asm volatile("s_wait_dscnt 0x0" ::: "memory"); // own-wave sP writes -> reads
    v16h pf0 = frag_a_load(&sP[w][0][0], 72);
    v16h pf1 = frag_a_load(&sP[w][0][32], 72);
#pragma unroll
    for (int dt = 0; dt < 8; ++dt) {
      acc[dt] = wmma_f16(pf0, frag_b_load(&sVt[dt * 16][0], 72), acc[dt]);
      acc[dt] = wmma_f16(pf1, frag_b_load(&sVt[dt * 16][32], 72), acc[dt]);
    }
  }

  // epilogue: O laid out (B*S, H*HD); uniform base + 32-bit offsets
  _Float16* obase = O + ((size_t)(b * 2048 + qbase)) * 4096 + h * 128;
  const int eoff = 8 * half * 4096 + ncol;
#pragma unroll
  for (int dt = 0; dt < 8; ++dt)
#pragma unroll
    for (int r = 0; r < 8; ++r)
      obase[eoff + r * 4096 + dt * 16] = (_Float16)(acc[dt][r] / rsum[r]);
}

// ---------------- orchestration ----------------
extern "C" void kernel_launch(void* const* d_in, const int* in_sizes, int n_in,
                              void* d_out, int out_size, void* d_ws, size_t ws_size,
                              hipStream_t stream) {
  const float* x    = (const float*)d_in[0];
  const float* wq   = (const float*)d_in[1];
  const float* wk   = (const float*)d_in[2];
  const float* wv   = (const float*)d_in[3];
  const float* wo   = (const float*)d_in[4];
  const float* fcos = (const float*)d_in[7];
  const float* fsin = (const float*)d_in[8];

  float* cache_k = (float*)d_out;                              // (B,S,KVH,HD)
  float* cache_v = cache_k + (size_t)2 * 2048 * 8 * 128;
  float* outp    = cache_v + (size_t)2 * 2048 * 8 * 128;       // (B,S,D)

  char* ws = (char*)d_ws;
  _Float16* xb     = (_Float16*)ws;                             // 32 MiB (aliased by attnb)
  _Float16* wqkvT  = (_Float16*)(ws + 33554432);                // 48 MiB, (6144 x 4096) N-major
  _Float16* woT    = (_Float16*)(ws + 33554432 + 50331648);     // 32 MiB, (4096 x 4096) N-major
  float*    qkvf   = (float*)(ws + 33554432 + 50331648 + 33554432); // 96 MiB
  _Float16* qb = wqkvT;                                         // alias: (B,H,S,HD) f16
  _Float16* kb = qb + (size_t)2 * 32 * 2048 * 128;              // (B,KVH,S,HD)
  _Float16* vb = kb + (size_t)2 * 8 * 2048 * 128;
  _Float16* attnb = xb;                                         // (B*S, H*HD)

  cvt_mat_kernel<<<16777216 / 256, 256, 0, stream>>>(x, xb, 4096, 4096, 4096, 0);
  cvt_transpose_kernel<<<dim3(4096 / 64, 4096 / 64), 256, 0, stream>>>(wq, wqkvT, 4096, 4096);
  cvt_transpose_kernel<<<dim3(1024 / 64, 4096 / 64), 256, 0, stream>>>(wk, wqkvT + (size_t)4096 * 4096, 4096, 1024);
  cvt_transpose_kernel<<<dim3(1024 / 64, 4096 / 64), 256, 0, stream>>>(wv, wqkvT + (size_t)5120 * 4096, 4096, 1024);
  cvt_transpose_kernel<<<dim3(4096 / 64, 4096 / 64), 256, 0, stream>>>(wo, woT, 4096, 4096);

  gemm_f16_kernel<<<dim3(6144 / 128, 4096 / 128), 256, 0, stream>>>(xb, wqkvT, qkvf, 4096, 6144, 4096);

  rope_q_kernel<<<8388608 / 256, 256, 0, stream>>>(qkvf, fcos, fsin, qb);
  rope_kv_kernel<<<2097152 / 256, 256, 0, stream>>>(qkvf, fcos, fsin, cache_k, cache_v, kb, vb);

  attn_kernel<<<dim3(32, 32, 2), 128, 0, stream>>>(qb, kb, vb, attnb);

  gemm_f16_kernel<<<dim3(4096 / 128, 4096 / 128), 256, 0, stream>>>(attnb, woT, outp, 4096, 4096, 4096);
}
